// Block_9844065042503
// MI455X (gfx1250) — compile-verified, matrix operand order checked by reading, not run on hardware
//
#include <hip/hip_runtime.h>
#include <hip/hip_bf16.h>
#include <math.h>

// ---------------------------------------------------------------------------
// FNet-style block on MI455X (gfx1250, wave32, WMMA).
//   f   = LN1(x) @ CosDFT          (Re(FFT) == GEMM with symmetric cos matrix)
//   x2  = x + LNf(f)
//   out = x2 + GELU(LN2(x2) @ w1 + b1) @ w2 + b2
// GEMMs: v_wmma_f32_16x16x32_f16 fed from double-buffered LDS tiles that are
// filled with GLOBAL_LOAD_ASYNC_TO_LDS_B128 (ASYNCcnt / s_wait_asynccnt).
// ---------------------------------------------------------------------------

typedef _Float16 h16v __attribute__((ext_vector_type(16)));
typedef _Float16 h8v  __attribute__((ext_vector_type(8)));
typedef float    f8v  __attribute__((ext_vector_type(8)));
typedef int      i4g  __attribute__((vector_size(16)));   // matches builtin proto

#define CDIM 1024
#define HDIM 4096
#define ROWS 8192   // B*N = 4*2048
#define LN_EPS 1e-5f

// ---- async global -> LDS 16B copy (per lane), CDNA5 ASYNCcnt path ---------
__device__ __forceinline__ void async_cp16(const _Float16* g, const _Float16* l)
{
#if __has_builtin(__builtin_amdgcn_global_load_async_to_lds_b128)
    __builtin_amdgcn_global_load_async_to_lds_b128(
        (__attribute__((address_space(1))) i4g*)(uintptr_t)g,
        (__attribute__((address_space(3))) i4g*)(unsigned)(uintptr_t)l,
        0, 0);
#else
    // VGLOBAL async form: VDST = LDS byte offset, VADDR = 64-bit global addr.
    asm volatile("global_load_async_to_lds_b128 %0, %1, off"
                 :: "v"((unsigned)(uintptr_t)l), "v"(g) : "memory");
#endif
}

__device__ __forceinline__ void async_wait0()
{
#if __has_builtin(__builtin_amdgcn_s_wait_asynccnt)
    __builtin_amdgcn_s_wait_asynccnt(0);
#else
    asm volatile("s_wait_asynccnt 0x0" ::: "memory");
#endif
}

// ---- WMMA fragment load (16-bit A/B layout, ISA 7.12.2) -------------------
// lane = {r = lane%16, h = lane/16}. For a row-major-in-K source row:
//   elements 0..7  <- K = 8h .. 8h+7 ; elements 8..15 <- K = 16+8h ..
__device__ __forceinline__ h16v load_frag(const _Float16* p) {
    h8v lo = *(const h8v*)(p);
    h8v hi = *(const h8v*)(p + 16);
    h16v r;
#pragma unroll
    for (int i = 0; i < 8; ++i) { r[i] = lo[i]; r[i + 8] = hi[i]; }
    return r;
}

// ---------------------------------------------------------------------------
// WMMA GEMM:  C[M,N] = A[M,K] (f16, row-major) x BT[N,K] (f16, "B^T").
// Block tile 64x128, 256 threads = 8 waves (4 row-waves x 2 col-waves),
// each wave owns a 16x64 strip = 4 accumulator tiles (v8f each).
// K-loop: 32 per step, LDS double buffer, async staging overlaps WMMA.
// Epilogues: 0 = store f32; 1 = +bias, exact GELU, store f16; 2 = +bias+resid.
// ---------------------------------------------------------------------------
#define LDSPAD 40   // 80B row pitch: 16B-aligned, spreads banks

__device__ __forceinline__ void stage_tiles(
    const _Float16* __restrict__ A, const _Float16* __restrict__ BT,
    int K, int bm0, int bn0, int kblk,
    _Float16 (*As)[LDSPAD], _Float16 (*Bs)[LDSPAD], int tid)
{
    {   // A tile 64x32 halfs = 256 x 16B chunks, one per thread
        const int row = tid >> 2, c = (tid & 3) * 8;
        async_cp16(A + (size_t)(bm0 + row) * K + kblk + c, &As[row][c]);
    }
#pragma unroll
    for (int i = 0; i < 2; ++i) {   // B tile 128x32 halfs = 512 chunks, 2/thread
        const int idx = tid + i * 256;
        const int row = idx >> 2, c = (idx & 3) * 8;
        async_cp16(BT + (size_t)(bn0 + row) * K + kblk + c, &Bs[row][c]);
    }
}

__global__ __launch_bounds__(256) void gemm_wmma_f16(
    const _Float16* __restrict__ A,
    const _Float16* __restrict__ BT,
    int M, int N, int K,
    const float* __restrict__ bias,
    const float* __restrict__ resid,
    float* __restrict__ outF,
    _Float16* __restrict__ outH,
    int ep)
{
    __shared__ __align__(16) _Float16 As[2][64][LDSPAD];
    __shared__ __align__(16) _Float16 Bs[2][128][LDSPAD];

    const int tid  = threadIdx.x;
    const int wave = tid >> 5;
    const int lane = tid & 31;
    const int r    = lane & 15;   // row (A) / col (B) within a 16x16 tile
    const int h    = lane >> 4;   // K-half selector

    const int wm  = wave & 3;                 // 4 row-waves
    const int wn  = wave >> 2;                // 2 col-waves
    const int bm0 = blockIdx.y * 64;
    const int bn0 = blockIdx.x * 128;

    f8v acc[4];
#pragma unroll
    for (int t = 0; t < 4; ++t)
#pragma unroll
        for (int j = 0; j < 8; ++j) acc[t][j] = 0.0f;

    stage_tiles(A, BT, K, bm0, bn0, 0, As[0], Bs[0], tid);
    async_wait0();
    __syncthreads();

    const int ksteps = K >> 5;
    for (int kb = 0; kb < ksteps; ++kb) {
        const int cur = kb & 1;
        if (kb + 1 < ksteps)   // prefetch next K-tile into the free buffer
            stage_tiles(A, BT, K, bm0, bn0, (kb + 1) << 5,
                        As[cur ^ 1], Bs[cur ^ 1], tid);

        h16v af = load_frag(&As[cur][wm * 16 + r][8 * h]);
#pragma unroll
        for (int t = 0; t < 4; ++t) {
            h16v bf = load_frag(&Bs[cur][wn * 64 + t * 16 + r][8 * h]);
            acc[t] = __builtin_amdgcn_wmma_f32_16x16x32_f16(
                false, af, false, bf, (short)0, acc[t], false, false);
        }

        async_wait0();     // our async writes into the next buffer are done
        __syncthreads();   // everyone done reading cur + sees staged data
    }

    // C/D layout: element j of v8f holds row (8h + j), col r of the tile.
#pragma unroll
    for (int t = 0; t < 4; ++t) {
        const int gn = bn0 + wn * 64 + t * 16 + r;
#pragma unroll
        for (int j = 0; j < 8; ++j) {
            const int gm = bm0 + wm * 16 + 8 * h + j;
            const size_t idx = (size_t)gm * N + gn;
            float v = acc[t][j];
            if (ep == 0) {
                outF[idx] = v;
            } else if (ep == 1) {
                v += bias[gn];
                float g = 0.5f * v * (1.0f + erff(v * 0.70710678118654752f));
                outH[idx] = (_Float16)g;
            } else {
                outF[idx] = v + bias[gn] + resid[idx];
            }
        }
    }
}

// ---------------------------------------------------------------------------
// LayerNorm over a 1024-wide row -> f16.  One 256-thread block per row.
// ---------------------------------------------------------------------------
__global__ __launch_bounds__(256) void ln_to_f16(
    const float* __restrict__ x,
    const float* __restrict__ g, const float* __restrict__ b,
    _Float16* __restrict__ out)
{
    __shared__ float s1[256], s2[256];
    const int row = blockIdx.x;
    const int tid = threadIdx.x;
    const float* xr = x + (size_t)row * CDIM;

    float v[4], s = 0.f, q = 0.f;
#pragma unroll
    for (int i = 0; i < 4; ++i) {
        v[i] = xr[tid + 256 * i];
        s += v[i]; q += v[i] * v[i];
    }
    s1[tid] = s; s2[tid] = q; __syncthreads();
    for (int off = 128; off > 0; off >>= 1) {
        if (tid < off) { s1[tid] += s1[tid + off]; s2[tid] += s2[tid + off]; }
        __syncthreads();
    }
    const float mu = s1[0] * (1.0f / CDIM);
    const float rs = rsqrtf(s2[0] * (1.0f / CDIM) - mu * mu + LN_EPS);
#pragma unroll
    for (int i = 0; i < 4; ++i) {
        const int c = tid + 256 * i;
        out[(size_t)row * CDIM + c] = (_Float16)((v[i] - mu) * rs * g[c] + b[c]);
    }
}

// ---------------------------------------------------------------------------
// x2 = x + LNf(f);  h2 = f16(LN2(x2)).  One block per row, two reductions.
// ---------------------------------------------------------------------------
__global__ __launch_bounds__(256) void lnf_residual_ln2(
    const float* __restrict__ x, const float* __restrict__ f,
    const float* __restrict__ lnf_g, const float* __restrict__ lnf_b,
    const float* __restrict__ ln2_g, const float* __restrict__ ln2_b,
    float* __restrict__ x2, _Float16* __restrict__ h2)
{
    __shared__ float s1[256], s2[256];
    const int row = blockIdx.x;
    const int tid = threadIdx.x;
    const size_t base = (size_t)row * CDIM;

    float fv[4], xv[4], s = 0.f, q = 0.f;
#pragma unroll
    for (int i = 0; i < 4; ++i) {
        const int c = tid + 256 * i;
        fv[i] = f[base + c];
        xv[i] = x[base + c];
        s += fv[i]; q += fv[i] * fv[i];
    }
    s1[tid] = s; s2[tid] = q; __syncthreads();
    for (int off = 128; off > 0; off >>= 1) {
        if (tid < off) { s1[tid] += s1[tid + off]; s2[tid] += s2[tid + off]; }
        __syncthreads();
    }
    float mu = s1[0] * (1.0f / CDIM);
    float rs = rsqrtf(s2[0] * (1.0f / CDIM) - mu * mu + LN_EPS);
    __syncthreads();   // before s1/s2 reuse

    float yv[4]; s = 0.f; q = 0.f;
#pragma unroll
    for (int i = 0; i < 4; ++i) {
        const int c = tid + 256 * i;
        yv[i] = xv[i] + (fv[i] - mu) * rs * lnf_g[c] + lnf_b[c];
        x2[base + c] = yv[i];
        s += yv[i]; q += yv[i] * yv[i];
    }
    s1[tid] = s; s2[tid] = q; __syncthreads();
    for (int off = 128; off > 0; off >>= 1) {
        if (tid < off) { s1[tid] += s1[tid + off]; s2[tid] += s2[tid + off]; }
        __syncthreads();
    }
    mu = s1[0] * (1.0f / CDIM);
    rs = rsqrtf(s2[0] * (1.0f / CDIM) - mu * mu + LN_EPS);
#pragma unroll
    for (int i = 0; i < 4; ++i) {
        const int c = tid + 256 * i;
        h2[base + c] = (_Float16)((yv[i] - mu) * rs * ln2_g[c] + ln2_b[c]);
    }
}

// ---------------------------------------------------------------------------
// One-time prep: cosine DFT matrix (symmetric -> row-major already "B^T"),
// and f32 -> f16 weight transposes.
// ---------------------------------------------------------------------------
__global__ void build_dft(_Float16* __restrict__ dftT)
{
    const unsigned idx = blockIdx.x * 256u + threadIdx.x;   // 1M entries
    const unsigned n = idx >> 10, k = idx & 1023u;
    const unsigned t = (n * k) & 1023u;
    dftT[idx] = (_Float16)cosf(6.283185307179586f * (float)t * (1.0f / 1024.0f));
}

__global__ void transpose_to_f16(const float* __restrict__ w,
                                 _Float16* __restrict__ wt,
                                 unsigned R, unsigned C)   // w:[R,C] -> wt:[C,R]
{
    const unsigned idx = blockIdx.x * 256u + threadIdx.x;
    if (idx >= R * C) return;
    const unsigned rr = idx / C, cc = idx % C;
    wt[(size_t)cc * R + rr] = (_Float16)w[idx];
}

// ---------------------------------------------------------------------------
extern "C" void kernel_launch(void* const* d_in, const int* in_sizes, int n_in,
                              void* d_out, int out_size, void* d_ws, size_t ws_size,
                              hipStream_t stream)
{
    (void)in_sizes; (void)n_in; (void)out_size; (void)ws_size;
    const float* x     = (const float*)d_in[0];
    const float* ln1_g = (const float*)d_in[1];
    const float* ln1_b = (const float*)d_in[2];
    const float* lnf_g = (const float*)d_in[3];
    const float* lnf_b = (const float*)d_in[4];
    const float* ln2_g = (const float*)d_in[5];
    const float* ln2_b = (const float*)d_in[6];
    const float* w1    = (const float*)d_in[7];
    const float* b1    = (const float*)d_in[8];
    const float* w2    = (const float*)d_in[9];
    const float* b2    = (const float*)d_in[10];
    float* out = (float*)d_out;

    // Workspace layout (bytes). m16 aliases fbuf (f is dead once x2/h2 exist).
    char* ws = (char*)d_ws;
    _Float16* dftT = (_Float16*)(ws);                         //  0 ..  2 MB
    _Float16* h1   = (_Float16*)(ws + (size_t)(2)  * 1048576);//  2 .. 18 MB
    float*    x2   = (float*)   (ws + (size_t)(18) * 1048576);// 18 .. 50 MB
    _Float16* h2   = (_Float16*)(ws + (size_t)(50) * 1048576);// 50 .. 66 MB
    _Float16* w1T  = (_Float16*)(ws + (size_t)(66) * 1048576);// 66 .. 74 MB
    _Float16* w2T  = (_Float16*)(ws + (size_t)(74) * 1048576);// 74 .. 82 MB
    float*    fbuf = (float*)   (ws + (size_t)(82) * 1048576);// 82 ..114 MB
    _Float16* m16  = (_Float16*)(ws + (size_t)(82) * 1048576);// 82 ..146 MB

    // --- prep: cos matrix + f16 transposed weights -------------------------
    build_dft<<<(CDIM * CDIM) / 256, 256, 0, stream>>>(dftT);
    transpose_to_f16<<<(CDIM * HDIM + 255) / 256, 256, 0, stream>>>(w1, w1T, CDIM, HDIM);
    transpose_to_f16<<<(HDIM * CDIM + 255) / 256, 256, 0, stream>>>(w2, w2T, HDIM, CDIM);

    // --- h1 = f16(LN1(x)) --------------------------------------------------
    ln_to_f16<<<ROWS, 256, 0, stream>>>(x, ln1_g, ln1_b, h1);

    // --- f = h1 @ Cos  (Re of per-token channel FFT) -----------------------
    gemm_wmma_f16<<<dim3(CDIM / 128, ROWS / 64), 256, 0, stream>>>(
        h1, dftT, ROWS, CDIM, CDIM, nullptr, nullptr, fbuf, nullptr, 0);

    // --- x2 = x + LNf(f);  h2 = f16(LN2(x2)) -------------------------------
    lnf_residual_ln2<<<ROWS, 256, 0, stream>>>(
        x, fbuf, lnf_g, lnf_b, ln2_g, ln2_b, x2, h2);

    // --- m = f16(GELU(h2 @ w1 + b1)) ---------------------------------------
    gemm_wmma_f16<<<dim3(HDIM / 128, ROWS / 64), 256, 0, stream>>>(
        h2, w1T, ROWS, HDIM, CDIM, b1, nullptr, nullptr, m16, 1);

    // --- out = x2 + m @ w2 + b2 --------------------------------------------
    gemm_wmma_f16<<<dim3(CDIM / 128, ROWS / 64), 256, 0, stream>>>(
        m16, w2T, ROWS, CDIM, HDIM, b2, x2, out, nullptr, 2);
}
